// EfficientMoE_31920196944054
// MI455X (gfx1250) — matrix operand order repaired and checked by reading, not compiled, not run on hardware
//
#include <hip/hip_runtime.h>
#include <stdint.h>

#define NTOK   8192      // B*T
#define C_DIM  1024
#define D_DIM  1024
#define NEXP   8
#define TOPK   2

#define BM 64
#define BN 128
#define BK 32
#define LDP (BK + 8)     // LDS row pitch in bf16: 40 (= 80 B, 16-B aligned subtiles)

typedef __attribute__((ext_vector_type(16))) __bf16 v16bf;
typedef __attribute__((ext_vector_type(8)))  __bf16 v8bf;
typedef __attribute__((ext_vector_type(2)))  __bf16 v2bf;
typedef __attribute__((ext_vector_type(8)))  float  v8f;

__device__ inline uint32_t bfbits(float f) {
    union { float f; uint32_t u; } v; v.f = f;
    return ((v.u + 0x7fffu + ((v.u >> 16) & 1u)) >> 16) & 0xffffu;  // RNE
}

// Pack two f32 -> packed bf16 pair in one VALU op where possible.
#if __has_builtin(__builtin_amdgcn_cvt_pk_bf16_f32)
__device__ inline uint32_t pack2(float lo, float hi) {
    v2bf r = __builtin_amdgcn_cvt_pk_bf16_f32(lo, hi);   // v_cvt_pk_bf16_f32 (RNE)
    union { v2bf v; uint32_t u; } c; c.v = r;
    return c.u;
}
#elif __has_builtin(__builtin_amdgcn_perm)
__device__ inline uint32_t pack2(float lo, float hi) {
    union { float f; uint32_t u; } a, b; a.f = lo; b.f = hi;
    return __builtin_amdgcn_perm(b.u, a.u, 0x07060302u); // v_perm_b32: {hi[31:16],lo[31:16]}
}
#else
__device__ inline uint32_t pack2(float lo, float hi) {
    return bfbits(lo) | (bfbits(hi) << 16);
}
#endif

// ---------------------------------------------------------------------------
// Kernel 0: zero output + per-expert counters (ws is poisoned by harness)
// ---------------------------------------------------------------------------
__global__ void moe_zero(float* __restrict__ out, int* __restrict__ counts) {
    size_t i = (size_t)blockIdx.x * blockDim.x + threadIdx.x;
    if (i < (size_t)NTOK * C_DIM) out[i] = 0.0f;
    if (i < NEXP) counts[i] = 0;
}

// ---------------------------------------------------------------------------
// Kernel 1: router. One wave32 per token: logits[8], top-2, softmax, scatter
// into per-expert token lists. Entry = (token << 1) | k_slot.
// ---------------------------------------------------------------------------
__global__ __launch_bounds__(256)
void moe_router(const float* __restrict__ x,
                const float* __restrict__ rw,     // [E, C]
                int*  __restrict__ counts,        // [E]
                int*  __restrict__ lists,         // [E, NTOK]
                float* __restrict__ probs)        // [NTOK, TOPK]
{
    const int tok  = (int)((blockIdx.x * blockDim.x + threadIdx.x) >> 5);
    const int lane = threadIdx.x & 31;
    if (tok >= NTOK) return;

    const float* xr = x + (size_t)tok * C_DIM;
    float acc[NEXP];
#pragma unroll
    for (int e = 0; e < NEXP; ++e) acc[e] = 0.0f;

    for (int c = lane; c < C_DIM; c += 32) {
        const float xv = xr[c];
#pragma unroll
        for (int e = 0; e < NEXP; ++e)
            acc[e] += xv * rw[e * C_DIM + c];
    }
#pragma unroll
    for (int e = 0; e < NEXP; ++e) {
        float a = acc[e];
#pragma unroll
        for (int off = 16; off > 0; off >>= 1)
            a += __shfl_xor(a, off, 32);
        acc[e] = a;
    }
    if (lane == 0) {
        int i0 = 0; float m0 = acc[0];
#pragma unroll
        for (int e = 1; e < NEXP; ++e) if (acc[e] > m0) { m0 = acc[e]; i0 = e; }
        int i1 = (i0 == 0) ? 1 : 0; float m1 = acc[i1];
#pragma unroll
        for (int e = 0; e < NEXP; ++e)
            if (e != i0 && acc[e] > m1) { m1 = acc[e]; i1 = e; }

        const float e1  = __expf(m1 - m0);
        const float inv = 1.0f / (1.0f + e1);

        probs[tok * TOPK + 0] = inv;
        probs[tok * TOPK + 1] = e1 * inv;
        int pos0 = atomicAdd(&counts[i0], 1);
        lists[i0 * NTOK + pos0] = (tok << 1) | 0;
        int pos1 = atomicAdd(&counts[i1], 1);
        lists[i1 * NTOK + pos1] = (tok << 1) | 1;
    }
}

// ---------------------------------------------------------------------------
// Fragment builders (16x16x32 bf16 WMMA; contiguous k = 16*kh + s, same order in A and B).
//  A tile in LDS row-major  lA[row][k]  (pitch LDP):
//    lane l: row = l&15; elements = lA[row][16*kh .. +16]   (two b128 loads)
//  B tile in LDS transposed lBt[n][k] (pitch LDP):
//    lane l: col = l&15; elements = lBt[col][16*kh .. +16]   (two b128 loads)
// ---------------------------------------------------------------------------
__device__ inline v16bf frag_a(const __bf16* rowp, int khalf) {
    v8bf a0 = *(const v8bf*)(rowp + 16 * khalf);
    v8bf a1 = *(const v8bf*)(rowp + 16 * khalf + 8);
    return __builtin_shufflevector(a0, a1, 0,1,2,3,4,5,6,7,8,9,10,11,12,13,14,15);
}
__device__ inline v16bf frag_b(const __bf16* colp, int khalf) {
    v8bf b0 = *(const v8bf*)(colp + 16 * khalf);
    v8bf b1 = *(const v8bf*)(colp + 16 * khalf + 8);
    return __builtin_shufflevector(b0, b1, 0,1,2,3,4,5,6,7,8,9,10,11,12,13,14,15);
}

// ---------------------------------------------------------------------------
// Kernel 2: hidden = relu(Xg @ W1_e)^2, grouped by expert, bf16 WMMA.
// grid = (NTOK/BM, NEXP, D/BN), block = 256 (8 waves, 4x2 wave tile).
// ---------------------------------------------------------------------------
__global__ __launch_bounds__(256)
void moe_gemm1(const float* __restrict__ x,       // [NTOK, C]
               const float* __restrict__ w1,      // [E, C, D]
               const int*   __restrict__ counts,
               const int*   __restrict__ lists,   // [E, NTOK]
               uint16_t*    __restrict__ hidden)  // [NTOK*TOPK, D] bf16
{
    const int e     = blockIdx.y;
    const int cnt   = counts[e];
    const int tileM = blockIdx.x;
    if (tileM * BM >= cnt) return;
    const int colBase = blockIdx.z * BN;

    __shared__ __bf16 lA [BM][LDP];
    __shared__ __bf16 lBt[BN][LDP];

    const int tid   = threadIdx.x;
    const int lane  = tid & 31;
    const int waveM = (tid >> 5) >> 1;   // 0..3
    const int waveN = (tid >> 5) & 1;    // 0..1
    const int khalf = lane >> 4;
    const int l15   = lane & 15;

    const int* mylist = lists + e * NTOK + tileM * BM;

    v8f acc[4];
#pragma unroll
    for (int s = 0; s < 4; ++s)
#pragma unroll
        for (int i = 0; i < 8; ++i) acc[s][i] = 0.0f;

    // A loader: thread -> row tid>>2 (0..63), 8 consecutive k (f32 -> bf16)
    const int  arow   = tid >> 2;
    const int  acol   = (tid & 3) * 8;
    const bool avalid = (tileM * BM + arow) < cnt;
    const int  atok   = avalid ? (mylist[arow] >> 1) : 0;
    const float* aptr = x + (size_t)atok * C_DIM + acol;

    // B loader: thread -> col n = tid&127, k-half = tid>>7 (16 k's, coalesced)
    const int bn  = tid & (BN - 1);
    const int bkh = tid >> 7;
    const float* bptr = w1 + ((size_t)e * C_DIM + bkh * 16) * D_DIM + colBase + bn;

    for (int k0 = 0; k0 < C_DIM; k0 += BK) {
        __builtin_prefetch(aptr + k0 + BK, 0, 3);
        __builtin_prefetch(bptr + (size_t)(k0 + BK) * D_DIM, 0, 3);

        // stage A: two float4 loads -> one 16-B packed store
        if (avalid) {
            const float4* ap = (const float4*)(aptr + k0);
            float4 v0 = ap[0], v1 = ap[1];
            uint4 st;
            st.x = pack2(v0.x, v0.y); st.y = pack2(v0.z, v0.w);
            st.z = pack2(v1.x, v1.y); st.w = pack2(v1.z, v1.w);
            *(uint4*)&lA[arow][acol] = st;
        } else {
            uint4 z; z.x = z.y = z.z = z.w = 0u;
            *(uint4*)&lA[arow][acol] = z;
        }
        // stage B transposed: 16 coalesced row reads -> 8 packed dwords
        {
            uint32_t pk[8];
#pragma unroll
            for (int i = 0; i < 8; ++i) {
                float f0 = bptr[(size_t)(k0 + 2 * i    ) * D_DIM];
                float f1 = bptr[(size_t)(k0 + 2 * i + 1) * D_DIM];
                pk[i] = pack2(f0, f1);
            }
            uint4* dst = (uint4*)&lBt[bn][bkh * 16];
            uint4 s0; s0.x = pk[0]; s0.y = pk[1]; s0.z = pk[2]; s0.w = pk[3];
            uint4 s1; s1.x = pk[4]; s1.y = pk[5]; s1.z = pk[6]; s1.w = pk[7];
            dst[0] = s0; dst[1] = s1;
        }
        __syncthreads();

        const v16bf afrag = frag_a(&lA[waveM * 16 + l15][0], khalf);
#pragma unroll
        for (int s = 0; s < 4; ++s) {
            const v16bf bfrag = frag_b(&lBt[waveN * 64 + s * 16 + l15][0], khalf);
            acc[s] = __builtin_amdgcn_wmma_f32_16x16x32_bf16(
                false, afrag, false, bfrag, (short)0, acc[s], false, false);
        }
        __syncthreads();
    }

    // epilogue: relu^2 -> bf16 hidden[(tok<<1)|k, colBase + ...]
#pragma unroll
    for (int s = 0; s < 4; ++s) {
        const int col = colBase + waveN * 64 + s * 16 + l15;
#pragma unroll
        for (int r = 0; r < 8; ++r) {
            const int lrow = waveM * 16 + r + 8 * khalf;
            const int grow = tileM * BM + lrow;
            if (grow < cnt) {
                const int entry = mylist[lrow];
                float v = acc[s][r];
                v = (v > 0.0f) ? v * v : 0.0f;
                hidden[(size_t)entry * D_DIM + col] = (uint16_t)bfbits(v);
            }
        }
    }
}

// ---------------------------------------------------------------------------
// Kernel 3: out[tok] += prob * (hidden_row @ W2_e). Same tiling; A operand is
// bf16 in memory -> staged via CDNA5 async global->LDS copy. grid =
// (NTOK/BM, NEXP, C/BN), block = 256.
// ---------------------------------------------------------------------------
__global__ __launch_bounds__(256)
void moe_gemm2(const uint16_t* __restrict__ hidden_raw, // [NTOK*TOPK, D] bf16
               const float*    __restrict__ w2,         // [E, D, C]
               const int*      __restrict__ counts,
               const int*      __restrict__ lists,
               const float*    __restrict__ probs,      // [NTOK, TOPK]
               float*          __restrict__ out)        // [NTOK, C]
{
    const __bf16* hidden = (const __bf16*)hidden_raw;
    const int e     = blockIdx.y;
    const int cnt   = counts[e];
    const int tileM = blockIdx.x;
    if (tileM * BM >= cnt) return;
    const int colBase = blockIdx.z * BN;

    __shared__ __bf16 lA [BM][LDP];
    __shared__ __bf16 lBt[BN][LDP];

    const int tid   = threadIdx.x;
    const int lane  = tid & 31;
    const int waveM = (tid >> 5) >> 1;
    const int waveN = (tid >> 5) & 1;
    const int khalf = lane >> 4;
    const int l15   = lane & 15;

    const int* mylist = lists + e * NTOK + tileM * BM;

    v8f acc[4];
#pragma unroll
    for (int s = 0; s < 4; ++s)
#pragma unroll
        for (int i = 0; i < 8; ++i) acc[s][i] = 0.0f;

    // A loader: 8 bf16 (16 B) per thread, already bf16 -> async copy to LDS
    const int  arow   = tid >> 2;
    const int  acol   = (tid & 3) * 8;
    const bool avalid = (tileM * BM + arow) < cnt;
    const int  aent   = avalid ? mylist[arow] : 0;
    const __bf16* aptr = hidden + (size_t)aent * D_DIM + acol;
    const uint32_t ldsdst = (uint32_t)(uintptr_t)&lA[arow][acol];

    const int bn  = tid & (BN - 1);
    const int bkh = tid >> 7;
    const float* bptr = w2 + ((size_t)e * D_DIM + bkh * 16) * C_DIM + colBase + bn;

    for (int k0 = 0; k0 < D_DIM; k0 += BK) {
        __builtin_prefetch(bptr + (size_t)(k0 + BK) * C_DIM, 0, 3);

        // stage A via CDNA5 async path: global_load_async_to_lds_b128
        if (avalid) {
            uint64_t ga = (uint64_t)(uintptr_t)(aptr + k0);
            asm volatile("global_load_async_to_lds_b128 %0, %1, off"
                         :: "v"(ldsdst), "v"(ga) : "memory");
        } else {
            uint4 z; z.x = z.y = z.z = z.w = 0u;
            *(uint4*)&lA[arow][acol] = z;
        }
        // stage B transposed (f32 -> packed bf16)
        {
            uint32_t pk[8];
#pragma unroll
            for (int i = 0; i < 8; ++i) {
                float f0 = bptr[(size_t)(k0 + 2 * i    ) * C_DIM];
                float f1 = bptr[(size_t)(k0 + 2 * i + 1) * C_DIM];
                pk[i] = pack2(f0, f1);
            }
            uint4* dst = (uint4*)&lBt[bn][bkh * 16];
            uint4 s0; s0.x = pk[0]; s0.y = pk[1]; s0.z = pk[2]; s0.w = pk[3];
            uint4 s1; s1.x = pk[4]; s1.y = pk[5]; s1.z = pk[6]; s1.w = pk[7];
            dst[0] = s0; dst[1] = s1;
        }
        asm volatile("s_wait_asynccnt 0" ::: "memory");
        __syncthreads();

        const v16bf afrag = frag_a(&lA[waveM * 16 + l15][0], khalf);
#pragma unroll
        for (int s = 0; s < 4; ++s) {
            const v16bf bfrag = frag_b(&lBt[waveN * 64 + s * 16 + l15][0], khalf);
            acc[s] = __builtin_amdgcn_wmma_f32_16x16x32_bf16(
                false, afrag, false, bfrag, (short)0, acc[s], false, false);
        }
        __syncthreads();
    }

    // epilogue: scale by router prob, scatter-add into out[tok]
#pragma unroll
    for (int s = 0; s < 4; ++s) {
        const int col = colBase + waveN * 64 + s * 16 + l15;
#pragma unroll
        for (int r = 0; r < 8; ++r) {
            const int lrow = waveM * 16 + r + 8 * khalf;
            const int grow = tileM * BM + lrow;
            if (grow < cnt) {
                const int entry = mylist[lrow];
                const int tok   = entry >> 1;
                const float p   = probs[entry];
                unsafeAtomicAdd(&out[(size_t)tok * C_DIM + col], p * acc[s][r]);
            }
        }
    }
}

// ---------------------------------------------------------------------------
extern "C" void kernel_launch(void* const* d_in, const int* in_sizes, int n_in,
                              void* d_out, int out_size, void* d_ws, size_t ws_size,
                              hipStream_t stream) {
    const float* x  = (const float*)d_in[0];   // [B,T,C]
    const float* rw = (const float*)d_in[1];   // [E,C]
    const float* w1 = (const float*)d_in[2];   // [E,C,D]
    const float* w2 = (const float*)d_in[3];   // [E,D,C]
    float* out = (float*)d_out;

    char* ws = (char*)d_ws;
    int*      counts = (int*)ws;                                        // 32 B
    int*      lists  = (int*)(ws + 256);                                // 256 KB
    float*    probs  = (float*)(ws + 256 + (size_t)NEXP * NTOK * 4);    // 64 KB
    uint16_t* hidden = (uint16_t*)(ws + 256 + (size_t)NEXP * NTOK * 4
                                      + (size_t)NTOK * TOPK * 4);       // 32 MB bf16

    const int zeroBlocks = (NTOK * C_DIM + 255) / 256;
    moe_zero<<<zeroBlocks, 256, 0, stream>>>(out, counts);

    moe_router<<<NTOK / 8, 256, 0, stream>>>(x, rw, counts, lists, probs);

    dim3 g1(NTOK / BM, NEXP, D_DIM / BN);
    moe_gemm1<<<g1, 256, 0, stream>>>(x, w1, counts, lists, hidden);

    dim3 g2(NTOK / BM, NEXP, C_DIM / BN);
    moe_gemm2<<<g2, 256, 0, stream>>>(hidden, w2, counts, lists, probs, out);
}
